// PointNet2Segmentor_71373766525331
// MI455X (gfx1250) — compile-verified, matrix operand order checked by reading, not run on hardware
//
#include <hip/hip_runtime.h>

// ---------------------------------------------------------------------------
// PointNet++ segmentor for MI455X (gfx1250).
// All MLP layers run as WMMA f16->f32 GEMMs (v_wmma_f32_16x16x32_f16).
// Weights are pre-swizzled into WMMA B-fragment order so every lane's B
// fragment is one aligned 32B vector load; each wave computes NT (4/8)
// N-tiles per A fragment for 4-8x A-traffic reuse. SA max-aggregation is
// fused into the last GEMM of each stage via float atomic max.
// ---------------------------------------------------------------------------

typedef __attribute__((ext_vector_type(16))) _Float16 v16h;
typedef __attribute__((ext_vector_type(8)))  float    v8f;

#define BATCH   4
#define NPTS    4096
#define N0PTS   2048
#define N1PTS   512
#define KNBR    64

// --------------------------- small helpers ---------------------------------

__device__ __forceinline__ void atomic_max_f32(float* addr, float val) {
  // sign-split trick: monotone over float ordering for both signs
  if (val >= 0.0f) atomicMax((int*)addr, __float_as_int(val));
  else             atomicMin((unsigned int*)addr, __float_as_uint(val));
}

__global__ void fill_f32_kernel(float* __restrict__ p, float v, int n) {
  int i = blockIdx.x * blockDim.x + threadIdx.x;
  if (i < n) p[i] = v;
}

// f32 row-major [Ks,N] -> f16 fragment-swizzled [tn][kc][lane][16] with K
// zero-padded to Kp. Element e of lane's v16h covers K = base(v,hi)+parity,
// matching the 16-bit A/B 16x32 WMMA VGPR layout.
__global__ void convert_w_frag_kernel(const float* __restrict__ src,
                                      _Float16* __restrict__ dst,
                                      int Ks, int Kp, int N) {
  int i = blockIdx.x * blockDim.x + threadIdx.x;
  if (i >= Kp * N) return;
  int e    = i & 15;
  int lane = (i >> 4) & 31;
  int kch  = Kp >> 5;
  int kc   = (i >> 9) % kch;
  int tn   = (i >> 9) / kch;
  int hi   = lane >> 4;
  int l15  = lane & 15;
  int v = e >> 1, p = e & 1;
  int kk  = ((v < 4) ? (2 * v) : (16 + 2 * (v - 4))) + 8 * hi + 32 * kc + p;
  int col = tn * 16 + l15;
  dst[i] = (kk < Ks) ? (_Float16)src[(size_t)kk * N + col] : (_Float16)0.0f;
}

// --------------------------- WMMA GEMM -------------------------------------
// C[M,N] = act(A[M,Kd] @ W[Kd,N] + bias). One wave per 16x(16*NT) tile strip.
// Wf is fragment-swizzled. OUTMODE 0: f16 store, 1: f32 store, 2: masked
// atomic-max into maxout (row valid iff nbr[row] >= 0, center = row >> 6).
template<int NT, int OUTMODE, bool RELU, bool HASBIAS>
__global__ __launch_bounds__(256) void wmma_gemm_kernel(
    const _Float16* __restrict__ A, const _Float16* __restrict__ Wf,
    const float* __restrict__ bias, void* __restrict__ out,
    float* __restrict__ maxout, const int* __restrict__ nbr,
    int M, int N, int Kd)
{
  int tilesM  = M >> 4;
  int groupsN = (N >> 4) / NT;
  int wid = blockIdx.x * (blockDim.x >> 5) + (threadIdx.x >> 5);
  if (wid >= tilesM * groupsN) return;     // whole-wave uniform: EXEC stays all-1s
  int tm = wid % tilesM;
  int gn = wid / tilesM;
  int lane = threadIdx.x & 31;
  int hi   = lane >> 4;                    // lane group (0: K 0..7/16..23, 1: K 8..15/24..31)
  int l15  = lane & 15;
  const _Float16* Arow = A + (size_t)(tm * 16 + l15) * Kd;
  int kchunks = Kd >> 5;

  v8f acc[NT] = {};
  for (int kc = 0; kc < kchunks; ++kc) {
    if (kc + 2 < kchunks)
      __builtin_prefetch(Arow + 32 * (kc + 2), 0, 1);   // global_prefetch_b8
    v16h a;
#pragma unroll
    for (int v = 0; v < 8; ++v) {
      int kk = ((v < 4) ? (2 * v) : (16 + 2 * (v - 4))) + 8 * hi + 32 * kc;
      a[2 * v]     = Arow[kk];
      a[2 * v + 1] = Arow[kk + 1];
    }
#pragma unroll
    for (int t = 0; t < NT; ++t) {
      int tn = gn * NT + t;
      const v16h* bp =
          (const v16h*)(Wf + (((size_t)tn * kchunks + kc) * 32 + lane) * 16);
      v16h b = *bp;                                     // 32B aligned vector load
      acc[t] = __builtin_amdgcn_wmma_f32_16x16x32_f16(false, a, false, b,
                                                      (short)0, acc[t],
                                                      false, false);
    }
  }

#pragma unroll
  for (int t = 0; t < NT; ++t) {
    int col = (gn * NT + t) * 16 + l15;
    float bb = 0.0f;
    if (HASBIAS) bb = bias[col];
#pragma unroll
    for (int r = 0; r < 8; ++r) {
      int orow = tm * 16 + 8 * hi + r;     // C layout: VGPR r -> M=r (+8 for hi lanes)
      float v = acc[t][r] + bb;
      if (RELU) v = fmaxf(v, 0.0f);
      if (OUTMODE == 0) {
        ((_Float16*)out)[(size_t)orow * N + col] = (_Float16)v;
      } else if (OUTMODE == 1) {
        ((float*)out)[(size_t)orow * N + col] = v;
      } else {
        if (nbr[orow] >= 0)
          atomic_max_f32(maxout + (size_t)(orow >> 6) * N + col, v);
      }
    }
  }
}

// --------------------------- FPS -------------------------------------------
// One block per batch. Replicates lax.scan semantics: emit last, update mind,
// next = argmax(mind) with lowest-index tie-break (matches jnp.argmax).
__global__ __launch_bounds__(256) void fps_kernel(
    const float* __restrict__ pos, int n, int m,
    int* __restrict__ idx, float* __restrict__ cpos)
{
  const float* P = pos + (size_t)blockIdx.x * n * 3;
  int*   I = idx  + (size_t)blockIdx.x * m;
  float* C = cpos + (size_t)blockIdx.x * m * 3;
  __shared__ float mind[NPTS];
  __shared__ float rv[256];
  __shared__ int   ri[256];
  __shared__ int   s_last;
  int t = threadIdx.x;
  for (int i = t; i < n; i += 256) mind[i] = 3.4e38f;
  if (t == 0) s_last = 0;
  __syncthreads();
  for (int it = 0; it < m; ++it) {
    int last = s_last;
    if (t == 0) {
      I[it] = last;
      C[it * 3 + 0] = P[last * 3 + 0];
      C[it * 3 + 1] = P[last * 3 + 1];
      C[it * 3 + 2] = P[last * 3 + 2];
    }
    float lx = P[last * 3], ly = P[last * 3 + 1], lz = P[last * 3 + 2];
    float bv = -1.0f; int bi = 0;
    for (int i = t; i < n; i += 256) {
      float dx = P[i * 3] - lx, dy = P[i * 3 + 1] - ly, dz = P[i * 3 + 2] - lz;
      float d  = dx * dx + dy * dy + dz * dz;
      float mv = fminf(mind[i], d);
      mind[i]  = mv;
      if (mv > bv) { bv = mv; bi = i; }
    }
    rv[t] = bv; ri[t] = bi;
    __syncthreads();
    for (int s = 128; s > 0; s >>= 1) {
      if (t < s) {
        if (rv[t + s] > rv[t] || (rv[t + s] == rv[t] && ri[t + s] < ri[t])) {
          rv[t] = rv[t + s]; ri[t] = ri[t + s];
        }
      }
      __syncthreads();
    }
    if (t == 0) s_last = ri[0];
    __syncthreads();
  }
}

// --------------------------- ball query ------------------------------------
// Exact K nearest within radius (matches top_k over masked -d2).
// grid = (m, B); one block per center. Unfilled slots = -1.
__global__ __launch_bounds__(256) void ball_query_kernel(
    const float* __restrict__ pos, const float* __restrict__ cpos,
    int n, int m, float r2, int* __restrict__ nbr)
{
  const float* P = pos  + (size_t)blockIdx.y * n * 3;
  const float* C = cpos + ((size_t)blockIdx.y * m + blockIdx.x) * 3;
  int* NB = nbr + ((size_t)blockIdx.y * m + blockIdx.x) * KNBR;
  __shared__ float d2[NPTS];
  __shared__ float rv[256];
  __shared__ int   ri[256];
  float cx = C[0], cy = C[1], cz = C[2];
  int t = threadIdx.x;
  for (int i = t; i < n; i += 256) {
    float dx = P[i * 3] - cx, dy = P[i * 3 + 1] - cy, dz = P[i * 3 + 2] - cz;
    float d  = dx * dx + dy * dy + dz * dz;
    d2[i] = (d <= r2) ? d : 3.4e38f;
  }
  __syncthreads();
  for (int k = 0; k < KNBR; ++k) {
    float bv = 3.4e38f; int bi = -1;
    for (int i = t; i < n; i += 256) {
      float d = d2[i];
      if (d < bv) { bv = d; bi = i; }
    }
    rv[t] = bv; ri[t] = bi;
    __syncthreads();
    for (int s = 128; s > 0; s >>= 1) {
      if (t < s) {
        if (rv[t + s] < rv[t] ||
            (rv[t + s] == rv[t] && ri[t + s] >= 0 &&
             (ri[t] < 0 || ri[t + s] < ri[t]))) {
          rv[t] = rv[t + s]; ri[t] = ri[t + s];
        }
      }
      __syncthreads();
    }
    int sel = (rv[0] <= r2) ? ri[0] : -1;
    if (t == 0) {
      NB[k] = sel;
      if (sel >= 0) d2[sel] = 3.4e38f;
    }
    __syncthreads();
  }
}

// --------------------------- feature builders ------------------------------

// SA0 rows: [x(6), pos_j - c(3), pad->32] f16
__global__ void build_feat0_kernel(const float* __restrict__ xb,
                                   const float* __restrict__ posb,
                                   const float* __restrict__ cposb,
                                   const int* __restrict__ nbrb,
                                   _Float16* __restrict__ feat, int rows) {
  int r = blockIdx.x * blockDim.x + threadIdx.x;
  if (r >= rows) return;
  int cm = r >> 6;
  int j  = nbrb[r];
  _Float16* F = feat + (size_t)r * 32;
#pragma unroll
  for (int c = 0; c < 32; ++c) F[c] = (_Float16)0.0f;
  if (j >= 0) {
    const float* X = xb + (size_t)j * 6;
#pragma unroll
    for (int c = 0; c < 6; ++c) F[c] = (_Float16)X[c];
    F[6] = (_Float16)(posb[j * 3 + 0] - cposb[cm * 3 + 0]);
    F[7] = (_Float16)(posb[j * 3 + 1] - cposb[cm * 3 + 1]);
    F[8] = (_Float16)(posb[j * 3 + 2] - cposb[cm * 3 + 2]);
  }
}

// SA1 rows: [x0(256), pos_j - c(3), pad->288] f16
__global__ void build_feat1_kernel(const float* __restrict__ x0b,
                                   const float* __restrict__ p0b,
                                   const float* __restrict__ p1b,
                                   const int* __restrict__ nbrb,
                                   _Float16* __restrict__ feat, int rows) {
  int r = blockIdx.x * blockDim.x + threadIdx.x;
  if (r >= rows) return;
  int cm = r >> 6;
  int j  = nbrb[r];
  _Float16* F = feat + (size_t)r * 288;
  if (j < 0) {
    for (int c = 0; c < 288; ++c) F[c] = (_Float16)0.0f;
    return;
  }
  const float* X = x0b + (size_t)j * 256;
  for (int c = 0; c < 256; ++c) F[c] = (_Float16)X[c];
  F[256] = (_Float16)(p0b[j * 3 + 0] - p1b[cm * 3 + 0]);
  F[257] = (_Float16)(p0b[j * 3 + 1] - p1b[cm * 3 + 1]);
  F[258] = (_Float16)(p0b[j * 3 + 2] - p1b[cm * 3 + 2]);
  for (int c = 259; c < 288; ++c) F[c] = (_Float16)0.0f;
}

// up0 input rows: [knn3-interp(x1)(1024), x0(256)] f16 (zero pos rows dropped)
__global__ void build_feat2_kernel(const float* __restrict__ x1,
                                   const float* __restrict__ pos1,
                                   const float* __restrict__ x0,
                                   const float* __restrict__ pos0,
                                   _Float16* __restrict__ feat) {
  int b = blockIdx.y, i = blockIdx.x;
  __shared__ int   si[3];
  __shared__ float sw[3];
  if (threadIdx.x == 0) {
    const float* P1 = pos1 + (size_t)b * N1PTS * 3;
    const float* P0 = pos0 + ((size_t)b * N0PTS + i) * 3;
    float px = P0[0], py = P0[1], pz = P0[2];
    float d[3] = {3.4e38f, 3.4e38f, 3.4e38f};
    int ix[3]  = {0, 0, 0};
    for (int j = 0; j < N1PTS; ++j) {
      float dx = px - P1[3 * j], dy = py - P1[3 * j + 1], dz = pz - P1[3 * j + 2];
      float dd = dx * dx + dy * dy + dz * dz;
      if (dd < d[0])      { d[2]=d[1]; ix[2]=ix[1]; d[1]=d[0]; ix[1]=ix[0]; d[0]=dd; ix[0]=j; }
      else if (dd < d[1]) { d[2]=d[1]; ix[2]=ix[1]; d[1]=dd; ix[1]=j; }
      else if (dd < d[2]) { d[2]=dd; ix[2]=j; }
    }
    float w0 = 1.0f / fmaxf(d[0], 1e-16f);
    float w1 = 1.0f / fmaxf(d[1], 1e-16f);
    float w2 = 1.0f / fmaxf(d[2], 1e-16f);
    float s  = w0 + w1 + w2;
    sw[0] = w0 / s; sw[1] = w1 / s; sw[2] = w2 / s;
    si[0] = ix[0];  si[1] = ix[1];  si[2] = ix[2];
  }
  __syncthreads();
  const float* X1 = x1 + (size_t)b * N1PTS * 1024;
  const float* A  = X1 + (size_t)si[0] * 1024;
  const float* Bp = X1 + (size_t)si[1] * 1024;
  const float* Cp = X1 + (size_t)si[2] * 1024;
  float w0 = sw[0], w1 = sw[1], w2 = sw[2];
  _Float16* F = feat + ((size_t)b * N0PTS + i) * 1280;
  for (int c = threadIdx.x; c < 1024; c += blockDim.x)
    F[c] = (_Float16)(w0 * A[c] + w1 * Bp[c] + w2 * Cp[c]);
  const float* X0 = x0 + ((size_t)b * N0PTS + i) * 256;
  for (int c = threadIdx.x; c < 256; c += blockDim.x)
    F[1024 + c] = (_Float16)X0[c];
}

// up1 input rows: [knn3-interp(x3)(256), x(6), pad->288] f16
__global__ void build_feat3_kernel(const float* __restrict__ x3,
                                   const float* __restrict__ pos0,
                                   const float* __restrict__ x,
                                   const float* __restrict__ pos,
                                   _Float16* __restrict__ feat) {
  int b = blockIdx.y, i = blockIdx.x;
  __shared__ int   si[3];
  __shared__ float sw[3];
  if (threadIdx.x == 0) {
    const float* P0 = pos0 + (size_t)b * N0PTS * 3;
    const float* P  = pos  + ((size_t)b * NPTS + i) * 3;
    float px = P[0], py = P[1], pz = P[2];
    float d[3] = {3.4e38f, 3.4e38f, 3.4e38f};
    int ix[3]  = {0, 0, 0};
    for (int j = 0; j < N0PTS; ++j) {
      float dx = px - P0[3 * j], dy = py - P0[3 * j + 1], dz = pz - P0[3 * j + 2];
      float dd = dx * dx + dy * dy + dz * dz;
      if (dd < d[0])      { d[2]=d[1]; ix[2]=ix[1]; d[1]=d[0]; ix[1]=ix[0]; d[0]=dd; ix[0]=j; }
      else if (dd < d[1]) { d[2]=d[1]; ix[2]=ix[1]; d[1]=dd; ix[1]=j; }
      else if (dd < d[2]) { d[2]=dd; ix[2]=j; }
    }
    float w0 = 1.0f / fmaxf(d[0], 1e-16f);
    float w1 = 1.0f / fmaxf(d[1], 1e-16f);
    float w2 = 1.0f / fmaxf(d[2], 1e-16f);
    float s  = w0 + w1 + w2;
    sw[0] = w0 / s; sw[1] = w1 / s; sw[2] = w2 / s;
    si[0] = ix[0];  si[1] = ix[1];  si[2] = ix[2];
  }
  __syncthreads();
  const float* X3 = x3 + (size_t)b * N0PTS * 256;
  const float* A  = X3 + (size_t)si[0] * 256;
  const float* Bp = X3 + (size_t)si[1] * 256;
  const float* Cp = X3 + (size_t)si[2] * 256;
  float w0 = sw[0], w1 = sw[1], w2 = sw[2];
  _Float16* F = feat + ((size_t)b * NPTS + i) * 288;
  for (int c = threadIdx.x; c < 256; c += blockDim.x)
    F[c] = (_Float16)(w0 * A[c] + w1 * Bp[c] + w2 * Cp[c]);
  if (threadIdx.x == 0) {
    const float* X = x + ((size_t)b * NPTS + i) * 6;
    for (int c = 0; c < 6; ++c)  F[256 + c] = (_Float16)X[c];
    for (int c = 262; c < 288; ++c) F[c] = (_Float16)0.0f;
  }
}

// last layer [64 -> 1], no bias, no relu
__global__ void final_layer_kernel(const _Float16* __restrict__ u2,
                                   const float* __restrict__ W4,
                                   float* __restrict__ out, int rows) {
  int r = blockIdx.x * blockDim.x + threadIdx.x;
  if (r >= rows) return;
  float s = 0.0f;
  for (int c = 0; c < 64; ++c) s += (float)u2[(size_t)r * 64 + c] * W4[c];
  out[r] = s;
}

// --------------------------- host orchestration ----------------------------

extern "C" void kernel_launch(void* const* d_in, const int* in_sizes, int n_in,
                              void* d_out, int out_size, void* d_ws, size_t ws_size,
                              hipStream_t stream) {
  (void)in_sizes; (void)n_in; (void)out_size; (void)ws_size;
  const float* x   = (const float*)d_in[0];   // [B,4096,6]
  const float* pos = (const float*)d_in[1];   // [B,4096,3]
  const float* sa0_w1 = (const float*)d_in[2];  const float* sa0_b1 = (const float*)d_in[3];
  const float* sa0_w2 = (const float*)d_in[4];  const float* sa0_b2 = (const float*)d_in[5];
  const float* sa0_w3 = (const float*)d_in[6];  const float* sa0_b3 = (const float*)d_in[7];
  const float* sa1_w1 = (const float*)d_in[8];  const float* sa1_b1 = (const float*)d_in[9];
  const float* sa1_w2 = (const float*)d_in[10]; const float* sa1_b2 = (const float*)d_in[11];
  const float* sa1_w3 = (const float*)d_in[12]; const float* sa1_b3 = (const float*)d_in[13];
  const float* up0_w1 = (const float*)d_in[14];
  const float* up0_w2 = (const float*)d_in[15];
  const float* up0_w3 = (const float*)d_in[16];
  const float* up1_w1 = (const float*)d_in[17];
  const float* up1_w2 = (const float*)d_in[18];
  const float* up1_w3 = (const float*)d_in[19];
  const float* up1_w4 = (const float*)d_in[20];

  // ---- bump allocator over d_ws ----
  char* ws = (char*)d_ws;
  size_t off = 0;
  auto alloc = [&](size_t bytes) -> void* {
    void* p = ws + off;
    off = (off + bytes + 255) & ~(size_t)255;
    return p;
  };

  // f16 fragment-swizzled weights (K padded to mult of 32; zero pos rows of
  // the up-stage first layers dropped since they multiply zeros)
  _Float16* wf_sa0_1 = (_Float16*)alloc((size_t)32 * 64 * 2);
  _Float16* wf_sa0_2 = (_Float16*)alloc((size_t)64 * 128 * 2);
  _Float16* wf_sa0_3 = (_Float16*)alloc((size_t)128 * 256 * 2);
  _Float16* wf_sa1_1 = (_Float16*)alloc((size_t)288 * 512 * 2);
  _Float16* wf_sa1_2 = (_Float16*)alloc((size_t)512 * 1024 * 2);
  _Float16* wf_sa1_3 = (_Float16*)alloc((size_t)1024 * 1024 * 2);
  _Float16* wf_up0_1 = (_Float16*)alloc((size_t)1280 * 1024 * 2);
  _Float16* wf_up0_2 = (_Float16*)alloc((size_t)1024 * 512 * 2);
  _Float16* wf_up0_3 = (_Float16*)alloc((size_t)512 * 256 * 2);
  _Float16* wf_up1_1 = (_Float16*)alloc((size_t)288 * 256 * 2);
  _Float16* wf_up1_2 = (_Float16*)alloc((size_t)256 * 128 * 2);
  _Float16* wf_up1_3 = (_Float16*)alloc((size_t)128 * 64 * 2);

  int*   idx0  = (int*)  alloc((size_t)BATCH * N0PTS * 4);
  float* cpos0 = (float*)alloc((size_t)BATCH * N0PTS * 3 * 4);
  int*   nbr0  = (int*)  alloc((size_t)BATCH * N0PTS * KNBR * 4);
  float* x0    = (float*)alloc((size_t)BATCH * N0PTS * 256 * 4);
  int*   idx1  = (int*)  alloc((size_t)BATCH * N1PTS * 4);
  float* cpos1 = (float*)alloc((size_t)BATCH * N1PTS * 3 * 4);
  int*   nbr1  = (int*)  alloc((size_t)BATCH * N1PTS * KNBR * 4);
  float* x1    = (float*)alloc((size_t)BATCH * N1PTS * 1024 * 4);
  float* x3    = (float*)alloc((size_t)BATCH * N0PTS * 256 * 4);

  _Float16* feat2 = (_Float16*)alloc((size_t)BATCH * N0PTS * 1280 * 2);
  _Float16* g0    = (_Float16*)alloc((size_t)BATCH * N0PTS * 1024 * 2);
  _Float16* g1    = (_Float16*)alloc((size_t)BATCH * N0PTS * 512 * 2);
  _Float16* feat3 = (_Float16*)alloc((size_t)BATCH * NPTS * 288 * 2);
  _Float16* u0    = (_Float16*)alloc((size_t)BATCH * NPTS * 256 * 2);
  _Float16* u1    = (_Float16*)alloc((size_t)BATCH * NPTS * 128 * 2);
  _Float16* u2    = (_Float16*)alloc((size_t)BATCH * NPTS * 64 * 2);

  // transient per-batch SA buffers (reused SA0/SA1)
  const size_t featT_elems = (size_t)N1PTS * KNBR * 288;   // >= SA0's 131072*32
  const size_t hA_elems    = (size_t)N1PTS * KNBR * 512;   // >= SA0's 131072*64
  const size_t hB_elems    = (size_t)N1PTS * KNBR * 1024;  // >= SA0's 131072*128
  _Float16* featT = (_Float16*)alloc(featT_elems * 2);
  _Float16* hA    = (_Float16*)alloc(hA_elems * 2);
  _Float16* hB    = (_Float16*)alloc(hB_elems * 2);

  auto cvt = [&](const float* src, _Float16* dst, int Ks, int Kp, int N) {
    int n = Kp * N;
    convert_w_frag_kernel<<<(n + 255) / 256, 256, 0, stream>>>(src, dst, Ks, Kp, N);
  };
  cvt(sa0_w1, wf_sa0_1,    9,   32,   64);
  cvt(sa0_w2, wf_sa0_2,   64,   64,  128);
  cvt(sa0_w3, wf_sa0_3,  128,  128,  256);
  cvt(sa1_w1, wf_sa1_1,  259,  288,  512);
  cvt(sa1_w2, wf_sa1_2,  512,  512, 1024);
  cvt(sa1_w3, wf_sa1_3, 1024, 1024, 1024);
  cvt(up0_w1, wf_up0_1, 1280, 1280, 1024);   // rows 1280..1282 multiply zeros
  cvt(up0_w2, wf_up0_2, 1024, 1024,  512);
  cvt(up0_w3, wf_up0_3,  512,  512,  256);
  cvt(up1_w1, wf_up1_1,  262,  288,  256);   // rows 262..264 multiply zeros
  cvt(up1_w2, wf_up1_2,  256,  256,  128);
  cvt(up1_w3, wf_up1_3,  128,  128,   64);

  auto gemm = [&](int outmode, bool relu, bool hasbias,
                  const _Float16* A, const _Float16* W, const float* bias,
                  void* out, float* maxout, const int* nbr,
                  int M, int N, int Kd) {
    int NT = (N >= 128) ? 8 : 4;
    int waves = (M >> 4) * ((N >> 4) / NT);
    dim3 grid((waves + 7) / 8);
    if (NT == 8) {
      if (outmode == 0 && relu && hasbias)
        wmma_gemm_kernel<8, 0, true, true><<<grid, 256, 0, stream>>>(A, W, bias, out, maxout, nbr, M, N, Kd);
      else if (outmode == 2)
        wmma_gemm_kernel<8, 2, false, true><<<grid, 256, 0, stream>>>(A, W, bias, out, maxout, nbr, M, N, Kd);
      else if (outmode == 0 && relu && !hasbias)
        wmma_gemm_kernel<8, 0, true, false><<<grid, 256, 0, stream>>>(A, W, bias, out, maxout, nbr, M, N, Kd);
      else
        wmma_gemm_kernel<8, 1, false, false><<<grid, 256, 0, stream>>>(A, W, bias, out, maxout, nbr, M, N, Kd);
    } else {
      if (hasbias)
        wmma_gemm_kernel<4, 0, true, true><<<grid, 256, 0, stream>>>(A, W, bias, out, maxout, nbr, M, N, Kd);
      else
        wmma_gemm_kernel<4, 0, true, false><<<grid, 256, 0, stream>>>(A, W, bias, out, maxout, nbr, M, N, Kd);
    }
  };

  // ---- SA0: FPS 4096->2048, ball query r=0.2, MLP [9,64,128,256], max ----
  fps_kernel<<<BATCH, 256, 0, stream>>>(pos, NPTS, N0PTS, idx0, cpos0);
  ball_query_kernel<<<dim3(N0PTS, BATCH), 256, 0, stream>>>(pos, cpos0, NPTS, N0PTS, 0.04f, nbr0);
  {
    int n = BATCH * N0PTS * 256;
    fill_f32_kernel<<<(n + 255) / 256, 256, 0, stream>>>(x0, -3.4e38f, n);
  }
  for (int b = 0; b < BATCH; ++b) {
    const int rows = N0PTS * KNBR;                    // 131072
    const float* xb = x   + (size_t)b * NPTS * 6;
    const float* pb = pos + (size_t)b * NPTS * 3;
    const float* cb = cpos0 + (size_t)b * N0PTS * 3;
    const int*   nb = nbr0  + (size_t)b * N0PTS * KNBR;
    float*       ob = x0    + (size_t)b * N0PTS * 256;
    build_feat0_kernel<<<(rows + 255) / 256, 256, 0, stream>>>(xb, pb, cb, nb, featT, rows);
    gemm(0, true,  true, featT, wf_sa0_1, sa0_b1, hA, nullptr, nullptr, rows, 64, 32);
    gemm(0, true,  true, hA,    wf_sa0_2, sa0_b2, hB, nullptr, nullptr, rows, 128, 64);
    gemm(2, false, true, hB,    wf_sa0_3, sa0_b3, nullptr, ob, nb,      rows, 256, 128);
  }

  // ---- SA1: FPS 2048->512, ball query r=0.4, MLP [259,512,1024,1024], max ----
  fps_kernel<<<BATCH, 256, 0, stream>>>(cpos0, N0PTS, N1PTS, idx1, cpos1);
  ball_query_kernel<<<dim3(N1PTS, BATCH), 256, 0, stream>>>(cpos0, cpos1, N0PTS, N1PTS, 0.16f, nbr1);
  {
    int n = BATCH * N1PTS * 1024;
    fill_f32_kernel<<<(n + 255) / 256, 256, 0, stream>>>(x1, -3.4e38f, n);
  }
  for (int b = 0; b < BATCH; ++b) {
    const int rows = N1PTS * KNBR;                    // 32768
    const float* x0b = x0    + (size_t)b * N0PTS * 256;
    const float* p0b = cpos0 + (size_t)b * N0PTS * 3;
    const float* p1b = cpos1 + (size_t)b * N1PTS * 3;
    const int*   nb  = nbr1  + (size_t)b * N1PTS * KNBR;
    float*       ob  = x1    + (size_t)b * N1PTS * 1024;
    build_feat1_kernel<<<(rows + 255) / 256, 256, 0, stream>>>(x0b, p0b, p1b, nb, featT, rows);
    gemm(0, true,  true, featT, wf_sa1_1, sa1_b1, hA, nullptr, nullptr, rows, 512, 288);
    gemm(0, true,  true, hA,    wf_sa1_2, sa1_b2, hB, nullptr, nullptr, rows, 1024, 512);
    gemm(2, false, true, hB,    wf_sa1_3, sa1_b3, nullptr, ob, nb,      rows, 1024, 1024);
  }

  // ---- up0: knn3 interp x1 -> level0, concat x0, MLP [1280,1024,512,256] ----
  build_feat2_kernel<<<dim3(N0PTS, BATCH), 128, 0, stream>>>(x1, cpos1, x0, cpos0, feat2);
  {
    const int rows = BATCH * N0PTS;                   // 8192
    gemm(0, true,  false, feat2, wf_up0_1, nullptr, g0, nullptr, nullptr, rows, 1024, 1280);
    gemm(0, true,  false, g0,    wf_up0_2, nullptr, g1, nullptr, nullptr, rows, 512, 1024);
    gemm(1, false, false, g1,    wf_up0_3, nullptr, x3, nullptr, nullptr, rows, 256, 512);
  }

  // ---- up1: knn3 interp x3 -> original pts, concat x, MLP [262,256,128,64,1] ----
  build_feat3_kernel<<<dim3(NPTS, BATCH), 64, 0, stream>>>(x3, cpos0, x, pos, feat3);
  {
    const int rows = BATCH * NPTS;                    // 16384
    gemm(0, true, false, feat3, wf_up1_1, nullptr, u0, nullptr, nullptr, rows, 256, 288);
    gemm(0, true, false, u0,    wf_up1_2, nullptr, u1, nullptr, nullptr, rows, 128, 256);
    gemm(0, true, false, u1,    wf_up1_3, nullptr, u2, nullptr, nullptr, rows, 64, 128);
    final_layer_kernel<<<(rows + 255) / 256, 256, 0, stream>>>(u2, up1_w4, (float*)d_out, rows);
  }
}
